// EmbeddingLoss_33767032881696
// MI455X (gfx1250) — matrix operand order repaired and checked by reading, not compiled, not run on hardware
//
#include <hip/hip_runtime.h>

typedef __attribute__((ext_vector_type(16))) _Float16 v16h;
typedef __attribute__((ext_vector_type(8)))  _Float16 v8h;
typedef __attribute__((ext_vector_type(8)))  float    v8f;

#define BS    2
#define IT    2
#define CH    32
#define NPIX  4096            // 64*64
#define TILES 256             // NPIX / 16
#define MT_PER_WAVE 32
#define CHUNKS 8              // TILES / MT_PER_WAVE
#define NWAVES (BS*IT*TILES*CHUNKS)   // 8192

// ---------------- Kernel 1: center + L2-normalize, f32 -> f16 ----------------
// emb layout: ((b*IT+i)*CH + c)*NPIX + n   -> eH layout: ((b*IT+i)*NPIX + n)*CH + c
__global__ __launch_bounds__(256) void normalize_kernel(
    const float* __restrict__ emb, _Float16* __restrict__ eH) {
  int p  = blockIdx.x * blockDim.x + threadIdx.x;   // 0 .. BS*IT*NPIX-1
  int n  = p & (NPIX - 1);
  int bi = p >> 12;                                  // NPIX = 2^12
  const float* src = emb + (size_t)bi * CH * NPIX + n;

  float v[CH];
  float mean = 0.0f;
#pragma unroll
  for (int c = 0; c < CH; ++c) { v[c] = src[(size_t)c * NPIX]; mean += v[c]; }
  mean *= (1.0f / CH);
  float ss = 0.0f;
#pragma unroll
  for (int c = 0; c < CH; ++c) { v[c] -= mean; ss += v[c] * v[c]; }
  float r = rsqrtf(ss);

  v16h lo, hi;
#pragma unroll
  for (int c = 0; c < 16; ++c) {
    lo[c] = (_Float16)(v[c]      * r);
    hi[c] = (_Float16)(v[c + 16] * r);
  }
  _Float16* dst = eH + (size_t)p * CH;               // 64-byte aligned
  *(v16h*)(dst)      = lo;
  *(v16h*)(dst + 16) = hi;
}

// ---------------- Kernel 2: label histogram -> 1/count weights --------------
__global__ __launch_bounds__(256) void weights_kernel(
    const long long* __restrict__ lab,
    float* __restrict__ preW, int* __restrict__ labi) {
  int b = blockIdx.x;
  __shared__ int hist[8];
  if (threadIdx.x < 8) hist[threadIdx.x] = 0;
  __syncthreads();
  for (int n = threadIdx.x; n < NPIX; n += blockDim.x) {
    int l = (int)lab[b * NPIX + n];
    atomicAdd(&hist[l], 1);
    labi[b * NPIX + n] = l;
  }
  __syncthreads();
  for (int n = threadIdx.x; n < NPIX; n += blockDim.x) {
    int l = labi[b * NPIX + n];
    preW[b * NPIX + n] = 1.0f / (float)hist[l];
  }
}

// ---------------- Kernel 3: WMMA Gram tiles + hinge loss --------------------
// One wave owns an n-tile (A fragment) and sweeps 32 m-tiles (B fragments).
// A 16-bit layout: lanes 0-15 hold K0..15 of row M=lane, lanes 16-31 K16..31.
// B 16-bit layout: lanes 0-15 hold K0..15 of column N=lane, lanes 16-31 K16..31.
// Loss simplification (sim = 0.5*d + 0.5):
//   pos: max(1-sim,0)   = 0.5*max(1-d, 0)
//   neg: max(sim-0.5,0) = 0.5*max(d,   0)
// 0.5 folded into final scale; wcol factored out of the weight product;
// hinge clamp emitted directly as v_max_num_f32 (avoids the compiler's
// fcanonicalize of the input — WMMA outputs are already quiet).
__device__ __forceinline__ float clamp0(float x) {
  float r;
  asm("v_max_num_f32_e32 %0, 0, %1" : "=v"(r) : "v"(x));
  return r;
}

__global__ __launch_bounds__(256) void gram_loss_kernel(
    const _Float16* __restrict__ eH,
    const float*    __restrict__ preW,
    const int*      __restrict__ labi,
    float*          __restrict__ part) {
  int lane  = threadIdx.x & 31;
  int wave  = blockIdx.x * (blockDim.x >> 5) + (threadIdx.x >> 5);
  int chunk = wave & (CHUNKS - 1);
  int tmp   = wave / CHUNKS;
  int tn    = tmp & (TILES - 1);
  int bi    = tmp >> 8;            // TILES = 2^8
  int b     = bi >> 1;             // IT = 2

  const _Float16* eBase = eH + (size_t)bi * NPIX * CH;
  int lane15 = lane & 15;
  int hi     = lane >> 4;          // 0 or 1

  // ---- A fragment (rows n0..n0+15), K contiguous, matching B ----
  int n0 = tn * 16;
  const _Float16* rowPtr = eBase + (size_t)(n0 + lane15) * CH;
  v8h a0 = *(const v8h*)(rowPtr + hi * 16);       // K = hi?16..23 : 0..7
  v8h a1 = *(const v8h*)(rowPtr + hi * 16 + 8);   // K = hi?24..31 : 8..15
  v16h afrag;
#pragma unroll
  for (int j = 0; j < 8; ++j) { afrag[j] = a0[j]; afrag[j + 8] = a1[j]; }

  // ---- per-row metadata (C/D layout: VGPR r -> row r + hi*8) ----
  float wrow[8]; int lrow[8];
#pragma unroll
  for (int r = 0; r < 8; ++r) {
    int row = n0 + r + hi * 8;
    wrow[r] = preW[b * NPIX + row];
    lrow[r] = labi[b * NPIX + row];
  }

  float acc = 0.0f;
  int mt0 = chunk * MT_PER_WAVE;
#pragma unroll 2
  for (int mt = mt0; mt < mt0 + MT_PER_WAVE; ++mt) {
    int m0 = mt * 16;
    // B fragment: lane = column, K contiguous (hi selects K 16..31)
    v16h bfrag = *(const v16h*)(eBase + (size_t)(m0 + lane15) * CH + hi * 16);
    int   col  = m0 + lane15;
    float wcol = preW[b * NPIX + col];
    int   lcol = labi[b * NPIX + col];

    v8f c = {};
    c = __builtin_amdgcn_wmma_f32_16x16x32_f16(
        /*neg_a=*/false, afrag, /*neg_b=*/false, bfrag,
        /*c_mod=*/(short)0, c, /*reuse_a=*/false, /*reuse_b=*/false);

    float t = 0.0f;
#pragma unroll
    for (int r = 0; r < 8; ++r) {
      float d   = c[r];
      float sel = (lrow[r] == lcol) ? (1.0f - d) : d;
      t = fmaf(wrow[r], clamp0(sel), t);
    }
    acc = fmaf(wcol, t, acc);
  }

  // wave32 reduction
#pragma unroll
  for (int s = 16; s > 0; s >>= 1) acc += __shfl_xor(acc, s, 32);
  if (lane == 0) part[wave] = acc;
}

// ---------------- Kernel 4: final reduction + scale -------------------------
__global__ __launch_bounds__(256) void reduce_kernel(
    const float* __restrict__ part, float* __restrict__ out) {
  float s = 0.0f;
  for (int i = threadIdx.x; i < NWAVES; i += 256) s += part[i];
  __shared__ float smem[8];
  int lane = threadIdx.x & 31, w = threadIdx.x >> 5;
#pragma unroll
  for (int o = 16; o > 0; o >>= 1) s += __shfl_xor(s, o, 32);
  if (lane == 0) smem[w] = s;
  __syncthreads();
  if (threadIdx.x == 0) {
    float t = 0.0f;
#pragma unroll
    for (int i = 0; i < 8; ++i) t += smem[i];
    // total = sum(weights*loss) * N/it, with the 0.5 from the hinge folded in:
    out[0] = t * (0.5f * (float)NPIX / (float)IT);   // = t * 1024
  }
}

// ---------------- Launcher ---------------------------------------------------
extern "C" void kernel_launch(void* const* d_in, const int* in_sizes, int n_in,
                              void* d_out, int out_size, void* d_ws, size_t ws_size,
                              hipStream_t stream) {
  const float*     emb = (const float*)d_in[0];       // (2,2,32,64,64) f32
  const long long* lab = (const long long*)d_in[1];   // (2,64,64) i64
  float*           out = (float*)d_out;               // scalar f32

  // workspace layout
  char* ws = (char*)d_ws;
  _Float16* eH   = (_Float16*)(ws);                                   // 1 MB
  float*    preW = (float*)(ws + (size_t)(1 << 20));                  // 32 KB
  int*      labi = (int*)(ws + (size_t)(1 << 20) + (32 << 10));       // 32 KB
  float*    part = (float*)(ws + (size_t)(1 << 20) + (64 << 10));     // 32 KB

  normalize_kernel<<<(BS * IT * NPIX) / 256, 256, 0, stream>>>(emb, eH);
  weights_kernel<<<BS, 256, 0, stream>>>(lab, preW, labi);
  gram_loss_kernel<<<NWAVES / 8, 256, 0, stream>>>(eH, preW, labi, part);
  reduce_kernel<<<1, 256, 0, stream>>>(part, out);
}
